// GateNetwork_51007031607839
// MI455X (gfx1250) — compile-verified, matrix operand order checked by reading, not run on hardware
//
#include <hip/hip_runtime.h>
#include <hip/hip_bf16.h>
#include <math.h>

typedef __attribute__((ext_vector_type(16))) __bf16 v16bf;
typedef __attribute__((ext_vector_type(16))) float  v16f;
typedef __attribute__((ext_vector_type(8)))  float  v8f;

#define K_DIM 3072
#define N_DIM 1024
#define NEXP  3

// native f32 -> bf16 (RNE) via hardware convert, bits for storage
static __device__ __forceinline__ unsigned short bf16_bits(float f) {
    return __builtin_bit_cast(unsigned short, (__bf16)f);
}
static __device__ __forceinline__ float bf16_to_f32(unsigned short s) {
    return __uint_as_float(((unsigned int)s) << 16);
}

// ---------------- Kernel 0: w1 [K,N] fp32 -> w1t [N,K] bf16 (6 MB, L2-resident) ---
__global__ __launch_bounds__(256)
void cvt_w1_kernel(const float* __restrict__ w1, unsigned short* __restrict__ w1t) {
    int idx = blockIdx.x * 256 + threadIdx.x;        // coalesced read over [K,N]
    int k = idx >> 10;                               // N_DIM == 1024
    int n = idx & (N_DIM - 1);
    w1t[(size_t)n * K_DIM + k] = bf16_bits(w1[idx]);
}

// ---------------- Kernel 1: h = GELU(x @ w1 + b1), bf16 WMMA, fp32 accum ----------
// Block = 256 threads (8 waves). Tile: M=128, N=128, K-step=32, double-buffered LDS.
// A tile: global fp32 -> regs -> v_cvt bf16 -> LDS (pipelined one K-step ahead).
// B tile: global bf16 -> LDS directly via global_load_async_to_lds_b128 (ASYNCcnt).
__global__ __launch_bounds__(256)
void gemm_gelu_kernel(const float* __restrict__ x,
                      const unsigned short* __restrict__ w1t,
                      const float* __restrict__ b1,
                      unsigned short* __restrict__ h,
                      int Btotal) {
    __shared__ alignas(64) unsigned short ldsA[2][128][32];   // [buf][M][K] 16 KB
    __shared__ alignas(64) unsigned short ldsB[2][128][32];   // [buf][N][K] 16 KB

    const int tid  = threadIdx.x;
    const int lane = tid & 31;
    const int wave = tid >> 5;
    const int wm   = wave >> 1;         // 0..3
    const int wn   = wave & 1;          // 0..1

    const int n0 = blockIdx.x * 128;    // N-tile fastest -> consecutive blocks share x panel (L2)
    const int m0 = blockIdx.y * 128;
    if (m0 >= Btotal) return;

    const int ldRow  = tid >> 1;        // 0..127
    const int ldHalf = (tid & 1) * 16;  // 0 or 16

    const float*          xPtr = x   + (size_t)(m0 + ldRow) * K_DIM + ldHalf;
    const unsigned short* wPtr = w1t + (size_t)(n0 + ldRow) * K_DIM + ldHalf;

    // ---- prologue: tile 0 in flight (x -> regs, w1t -> LDS async)
    float4 xf[4];
    {
        const float4* p = reinterpret_cast<const float4*>(xPtr);
        xf[0] = p[0]; xf[1] = p[1]; xf[2] = p[2]; xf[3] = p[3];
    }
    {
        unsigned dst = (unsigned)(size_t)&ldsB[0][ldRow][ldHalf];
        const unsigned short* src = wPtr;
        asm volatile("global_load_async_to_lds_b128 %0, %1, off\n\t"
                     "global_load_async_to_lds_b128 %0, %1, off offset:16"
                     :: "v"(dst), "v"(src) : "memory");
    }

    v8f acc[2][4];
#pragma unroll
    for (int a = 0; a < 2; ++a)
#pragma unroll
        for (int b = 0; b < 4; ++b)
#pragma unroll
            for (int r = 0; r < 8; ++r) acc[a][b][r] = 0.0f;

    const int l16 = lane & 15;
    const int lh  = (lane >> 4) * 16;

    for (int kk = 0; kk < K_DIM; kk += 32) {
        const int buf = (kk >> 5) & 1;

        // ---- stage A: hardware packed f32->bf16 convert, one 32B LDS store
        v16f xv;
        xv[0]  = xf[0].x; xv[1]  = xf[0].y; xv[2]  = xf[0].z; xv[3]  = xf[0].w;
        xv[4]  = xf[1].x; xv[5]  = xf[1].y; xv[6]  = xf[1].z; xv[7]  = xf[1].w;
        xv[8]  = xf[2].x; xv[9]  = xf[2].y; xv[10] = xf[2].z; xv[11] = xf[2].w;
        xv[12] = xf[3].x; xv[13] = xf[3].y; xv[14] = xf[3].z; xv[15] = xf[3].w;
        v16bf xb = __builtin_convertvector(xv, v16bf);
        *reinterpret_cast<v16bf*>(&ldsA[buf][ldRow][ldHalf]) = xb;

        // B tile for this buffer must have landed in LDS
        asm volatile("s_wait_asynccnt 0x0" ::: "memory");
        __syncthreads();

        // ---- issue next tile (overlaps with WMMA below); post-barrier => race-free
        if (kk + 32 < K_DIM) {
            const float4* p = reinterpret_cast<const float4*>(xPtr + kk + 32);
            xf[0] = p[0]; xf[1] = p[1]; xf[2] = p[2]; xf[3] = p[3];
            unsigned dst = (unsigned)(size_t)&ldsB[buf ^ 1][ldRow][ldHalf];
            const unsigned short* src = wPtr + kk + 32;
            asm volatile("global_load_async_to_lds_b128 %0, %1, off\n\t"
                         "global_load_async_to_lds_b128 %0, %1, off offset:16"
                         :: "v"(dst), "v"(src) : "memory");
        }

        // ---- fragments, contiguous k order used by this code for A and B: lane m holds K0..15, lane m+16 holds K16..31
        v16bf Af[2], Bf[4];
#pragma unroll
        for (int a = 0; a < 2; ++a)
            Af[a] = *reinterpret_cast<const v16bf*>(&ldsA[buf][wm * 32 + a * 16 + l16][lh]);
#pragma unroll
        for (int b = 0; b < 4; ++b)
            Bf[b] = *reinterpret_cast<const v16bf*>(&ldsB[buf][wn * 64 + b * 16 + l16][lh]);

#pragma unroll
        for (int a = 0; a < 2; ++a)
#pragma unroll
            for (int b = 0; b < 4; ++b)
                acc[a][b] = __builtin_amdgcn_wmma_f32_16x16x32_bf16(
                    false, Af[a], false, Bf[b], (short)0, acc[a][b], false, false);
    }

    // ---- epilogue: + b1, exact GELU, store bf16 h
    const int lh8 = (lane >> 4) * 8;
#pragma unroll
    for (int a = 0; a < 2; ++a) {
#pragma unroll
        for (int b = 0; b < 4; ++b) {
            const int   ncol  = n0 + wn * 64 + b * 16 + l16;
            const float bias  = b1[ncol];
            const int   mbase = m0 + wm * 32 + a * 16 + lh8;
#pragma unroll
            for (int r = 0; r < 8; ++r) {
                float v = acc[a][b][r] + bias;
                float g = 0.5f * v * (1.0f + erff(v * 0.70710678118654752f));
                h[(size_t)(mbase + r) * N_DIM + ncol] = bf16_bits(g);
            }
        }
    }
}

// ---------------- Kernel 2: gates = softmax(h@w2+b2); stable top-k mask; renorm ---
__global__ __launch_bounds__(256)
void gate_kernel(const unsigned short* __restrict__ h,
                 const float* __restrict__ w2,
                 const float* __restrict__ b2,
                 const int* __restrict__ kptr,
                 float* __restrict__ out_gated,
                 float* __restrict__ out_mask,
                 int Btotal) {
    __shared__ float w2s[N_DIM * NEXP];
    for (int i = threadIdx.x; i < N_DIM * NEXP; i += 256) w2s[i] = w2[i];
    __syncthreads();

    const int lane = threadIdx.x & 31;
    const int wave = threadIdx.x >> 5;
    const int row  = blockIdx.x * 8 + wave;
    if (row >= Btotal) return;

    const unsigned short* hr = h + (size_t)row * N_DIM;
    float s0 = 0.f, s1 = 0.f, s2 = 0.f;
    for (int c = lane; c < N_DIM; c += 32) {       // coalesced b16 loads across wave32
        float v = bf16_to_f32(hr[c]);
        s0 = fmaf(v, w2s[c * 3 + 0], s0);
        s1 = fmaf(v, w2s[c * 3 + 1], s1);
        s2 = fmaf(v, w2s[c * 3 + 2], s2);
    }
#pragma unroll
    for (int off = 16; off > 0; off >>= 1) {       // wave32 butterfly reduction
        s0 += __shfl_xor(s0, off, 32);
        s1 += __shfl_xor(s1, off, 32);
        s2 += __shfl_xor(s2, off, 32);
    }
    if (lane == 0) {
        float g0 = s0 + b2[0], g1 = s1 + b2[1], g2 = s2 + b2[2];
        float mx = fmaxf(g0, fmaxf(g1, g2));
        float e0 = __expf(g0 - mx), e1 = __expf(g1 - mx), e2 = __expf(g2 - mx);
        float inv = 1.0f / (e0 + e1 + e2);
        float p0 = e0 * inv, p1 = e1 * inv, p2 = e2 * inv;
        int kk = kptr[0]; kk = kk < 0 ? 0 : (kk > NEXP ? NEXP : kk);
        // stable top-k rank (earlier index wins ties, matching lax.top_k)
        int r0 = (p1 >  p0) + (p2 >  p0);
        int r1 = (p0 >= p1) + (p2 >  p1);
        int r2 = (p0 >= p2) + (p1 >= p2);
        float m0v = (r0 < kk) ? 1.f : 0.f;
        float m1v = (r1 < kk) ? 1.f : 0.f;
        float m2v = (r2 < kk) ? 1.f : 0.f;
        float q0 = p0 * m0v, q1 = p1 * m1v, q2 = p2 * m2v;
        float d = 1.0f / (q0 + q1 + q2 + 1e-8f);
        out_gated[row * 3 + 0] = q0 * d;
        out_gated[row * 3 + 1] = q1 * d;
        out_gated[row * 3 + 2] = q2 * d;
        out_mask[row * 3 + 0] = m0v;
        out_mask[row * 3 + 1] = m1v;
        out_mask[row * 3 + 2] = m2v;
    }
}

extern "C" void kernel_launch(void* const* d_in, const int* in_sizes, int n_in,
                              void* d_out, int out_size, void* d_ws, size_t ws_size,
                              hipStream_t stream) {
    const float* x  = (const float*)d_in[0];
    const float* w1 = (const float*)d_in[1];
    const float* b1 = (const float*)d_in[2];
    const float* w2 = (const float*)d_in[3];
    const float* b2 = (const float*)d_in[4];
    const int*   kp = (const int*)d_in[5];
    const int Btotal = in_sizes[0] / K_DIM;

    // workspace: [w1t bf16: 6 MB][h bf16: B*1024*2 = 128 MB]
    unsigned short* w1t = (unsigned short*)d_ws;
    unsigned short* h   = (unsigned short*)((char*)d_ws +
                          (size_t)N_DIM * K_DIM * sizeof(unsigned short));

    cvt_w1_kernel<<<(K_DIM * N_DIM) / 256, 256, 0, stream>>>(w1, w1t);

    dim3 g1(N_DIM / 128, (Btotal + 127) / 128);   // N fastest -> x-panel L2 reuse
    gemm_gelu_kernel<<<g1, 256, 0, stream>>>(x, w1t, b1, h, Btotal);

    float* out_gated = (float*)d_out;
    float* out_mask  = out_gated + (size_t)Btotal * NEXP;
    gate_kernel<<<(Btotal + 7) / 8, 256, 0, stream>>>(h, w2, b2, kp,
                                                      out_gated, out_mask, Btotal);
}
